// VAE_987842478703
// MI455X (gfx1250) — matrix operand order repaired and checked by reading, not compiled, not run on hardware
//
#include <hip/hip_runtime.h>
#include <hip/hip_bf16.h>
#include <cstdint>
#include <cstddef>

// ---------------------------------------------------------------------------
// Types for CDNA5 WMMA (wave32): v_wmma_f32_16x16x32_f16
// ---------------------------------------------------------------------------
typedef __attribute__((ext_vector_type(16))) _Float16 v16h_t;
typedef __attribute__((ext_vector_type(8)))  _Float16 v8h_t;
typedef __attribute__((ext_vector_type(8)))  float    v8f_t;

#define WGSZ 256  // 8 wave32 waves per workgroup

__device__ inline v16h_t mk16(v8h_t lo, v8h_t hi) {
  v16h_t a;
#pragma unroll
  for (int i = 0; i < 8; ++i) { a[i] = lo[i]; a[i + 8] = hi[i]; }
  return a;
}

__device__ inline v8f_t wmma16(v16h_t a, v16h_t b, v8f_t c) {
  // 8 args: (neg_a, A, neg_b, B, c_mod, C, reuse_a, reuse_b)
  return __builtin_amdgcn_wmma_f32_16x16x32_f16(false, a, false, b, (short)0, c,
                                                false, false);
}

// ---------------------------------------------------------------------------
// Fast transcendentals: the gate math sits on the serial recurrent critical
// path, so use raw HW trans ops (v_exp_f32 / v_rcp_f32 / v_tanh_f32) instead
// of libm range-reduction + IEEE division sequences.
// ---------------------------------------------------------------------------
#define LOG2E_F 1.4426950408889634f

__device__ inline float hsig_f(float x) {
  float y = 0.2f * x + 0.5f;
  return fminf(fmaxf(y, 0.0f), 1.0f);
}

__device__ inline float fast_sigmoid(float x) {
  // 1/(1+e^-x) = rcp(1 + 2^(-x*log2e));  x->-inf: rcp(inf)=0, x->+inf: rcp(1)=1
  return __builtin_amdgcn_rcpf(1.0f + __builtin_amdgcn_exp2f(-LOG2E_F * x));
}

__device__ inline float fast_tanh(float x) {
#if __has_builtin(__builtin_amdgcn_tanhf)
  return __builtin_amdgcn_tanhf(x);  // v_tanh_f32 (CDNA5 TRANS op)
#else
  // tanh(x) = 1 - 2/(2^(2x*log2e)+1);  saturates correctly at +-inf
  float e = __builtin_amdgcn_exp2f(2.0f * LOG2E_F * x);
  return 1.0f - 2.0f * __builtin_amdgcn_rcpf(e + 1.0f);
#endif
}

// ---------------------------------------------------------------------------
// Pad x (B,T,4) f32 -> Xp (B*T,32) f16, and sequence mask = any(x!=0)
// ---------------------------------------------------------------------------
__global__ void vae_pad_x(const float* __restrict__ x, _Float16* __restrict__ xp,
                          float* __restrict__ maskb, int BT) {
  int i = blockIdx.x * WGSZ + threadIdx.x;
  if (i >= BT) return;
  const float* xr = x + (size_t)i * 4;
  float a0 = xr[0], a1 = xr[1], a2 = xr[2], a3 = xr[3];
  maskb[i] = (a0 != 0.f || a1 != 0.f || a2 != 0.f || a3 != 0.f) ? 1.f : 0.f;
  _Float16* o = xp + (size_t)i * 32;
  o[0] = (_Float16)a0; o[1] = (_Float16)a1; o[2] = (_Float16)a2; o[3] = (_Float16)a3;
#pragma unroll
  for (int c = 4; c < 32; ++c) o[c] = (_Float16)0.f;
}

// ---------------------------------------------------------------------------
// Repack a (Krows x 3H) f32 weight into WMMA B-fragment swizzled f16 layout.
// frag f = nt*ksteps + ks; element (lane,i) holds W[k][g*H+h] with
//   k = ks*32 + (lane>>4)*16 + i,  n = nt*16 + (lane&15), g = n/Hp, h = n%Hp
// ---------------------------------------------------------------------------
__global__ void vae_repack_frag(const float* __restrict__ W, int Krows, int H,
                                int Hp, int ksteps, int ntiles,
                                _Float16* __restrict__ out) {
  int idx = blockIdx.x * WGSZ + threadIdx.x;
  int total = ntiles * ksteps * 512;
  if (idx >= total) return;
  int f = idx >> 9;
  int r = idx & 511;
  int lane = r >> 4;
  int i = r & 15;
  int ks = f % ksteps;
  int nt = f / ksteps;
  int k = ks * 32 + ((lane >> 4) << 4) + i;
  int n = nt * 16 + (lane & 15);
  int g = n / Hp;
  int h = n - g * Hp;
  _Float16 v = (_Float16)0.f;
  if (k < Krows && h < H) v = (_Float16)W[(size_t)k * (3 * H) + g * H + h];
  out[idx] = v;
}

// b (2,3H) f32 -> padded input/recurrent biases (3*Hp each)
__global__ void vae_repack_bias(const float* __restrict__ b, int H, int Hp,
                                float* __restrict__ b0p, float* __restrict__ brp) {
  int idx = blockIdx.x * WGSZ + threadIdx.x;
  if (idx >= 3 * Hp) return;
  int g = idx / Hp;
  int h = idx - g * Hp;
  float v0 = 0.f, v1 = 0.f;
  if (h < H) { v0 = b[g * H + h]; v1 = b[3 * H + g * H + h]; }
  b0p[idx] = v0;
  brp[idx] = v1;
}

// ---------------------------------------------------------------------------
// Input projection GEMM: C[M x Ntot] = A[M x (KSTEPS*32)] (f16) @ Bsw + bias
// One wave per 16-row tile; A fragments cached in VGPRs across the N loop.
// ---------------------------------------------------------------------------
template <int KSTEPS>
__global__ void vae_gemm_xp(const _Float16* __restrict__ A,
                            const _Float16* __restrict__ Bsw,
                            const float* __restrict__ bias,
                            float* __restrict__ C, int M, int Ntot, int ntiles) {
  int lane = threadIdx.x & 31;
  int wid = threadIdx.x >> 5;
  int mt = blockIdx.x * 8 + wid;
  if (mt * 16 >= M) return;
  int kb = ((lane >> 4) << 4);
  const _Float16* arow = A + (size_t)(mt * 16 + (lane & 15)) * (KSTEPS * 32);

  v16h_t af[KSTEPS];
#pragma unroll
  for (int ks = 0; ks < KSTEPS; ++ks) {
    v8h_t lo = *(const v8h_t*)(arow + ks * 32 + kb);
    v8h_t hi = *(const v8h_t*)(arow + ks * 32 + kb + 8);
    af[ks] = mk16(lo, hi);
  }

  int col = (lane & 15);
  int mrow0 = mt * 16 + ((lane >> 4) << 3);
  for (int nt = 0; nt < ntiles; ++nt) {
    float bv = bias[nt * 16 + col];
    v8f_t acc = {bv, bv, bv, bv, bv, bv, bv, bv};
#pragma unroll
    for (int ks = 0; ks < KSTEPS; ++ks) {
      v16h_t bf = *(const v16h_t*)(Bsw + (size_t)(nt * KSTEPS + ks) * 512 + lane * 16);
      acc = wmma16(af[ks], bf, acc);
    }
    size_t base = (size_t)mrow0 * Ntot + nt * 16 + col;
#pragma unroll
    for (int r = 0; r < 8; ++r) C[base + (size_t)r * Ntot] = acc[r];
  }
}

// ---------------------------------------------------------------------------
// GRU recurrence, templated on UKS (k-steps over the padded state dim):
//   Hp = 32*UKS, Ntot = 3*Hp, ntiles = Ntot/16, elems/thread = 2*UKS (exact).
// One workgroup per 16-row batch tile (8 total). U resident in LDS as
// swizzled WMMA fragments; f16 h-state in LDS; 2 barriers per step.
// Per step:
//   phase 0: issue global xp loads for this step into registers (latency
//            overlapped with phase 1) + prefetch next step's lines
//   phase 1: A-frags (h-state) loaded ONCE into regs, then WMMA over n-tiles
//   phase 2: fuse gates (HW trans ops), update h-state
// actmode 0: hard_sigmoid + tanh ; 1: sigmoid + identity
// ---------------------------------------------------------------------------
template <int UKS>
__global__ void vae_gru_recur(const float* __restrict__ xp,      // (B*T) x Ntot
                              const _Float16* __restrict__ uswg, // swizzled U
                              const float* __restrict__ brp,     // Ntot
                              const float* __restrict__ maskb,   // B*T or null
                              _Float16* __restrict__ hseq,       // (B*T)xHp or null
                              float* __restrict__ hfin,          // B x Hp or null
                              int T, int actmode) {
  constexpr int Hp = 32 * UKS;
  constexpr int Ntot = 3 * Hp;
  constexpr int NT = Ntot / 16;
  constexpr int EPT = 2 * UKS;  // elements per thread in phase 2 (exact)

  extern __shared__ char smem[];
  _Float16* usw = (_Float16*)smem;                  // NT*UKS*512 halves
  _Float16* hl  = usw + (size_t)NT * UKS * 512;     // 16*Hp
  float* stage  = (float*)(hl + 16 * Hp);           // 16*Ntot
  float* brl    = stage + 16 * Ntot;                // Ntot

  int tid = threadIdx.x;
  int lane = tid & 31;
  int wid = tid >> 5;
  int b0 = blockIdx.x * 16;

  // One-time fills (vectorized 16B copies for U)
  {
    const uint4* src = (const uint4*)uswg;
    uint4* dst = (uint4*)usw;
    int n16 = NT * UKS * 512 / 8;  // halves -> uint4 count
    for (int i = tid; i < n16; i += WGSZ) dst[i] = src[i];
    for (int i = tid; i < 16 * Hp; i += WGSZ) hl[i] = (_Float16)0.f;
    for (int i = tid; i < Ntot; i += WGSZ) brl[i] = brp[i];
  }
  __syncthreads();

  int kb = ((lane >> 4) << 4);
  int col = lane & 15;
  int mb = (lane >> 4) * 8;
  const _Float16* arow = hl + (size_t)col * Hp;

  for (int t = 0; t < T; ++t) {
    // ---- phase 0: issue this step's global xp loads; prefetch t+1 ----
    float xzv[EPT], xrv[EPT], xhv[EPT], mtv[EPT];
#pragma unroll
    for (int j = 0; j < EPT; ++j) {
      int e = tid + j * WGSZ;
      int m = e / Hp;
      int h = e - m * Hp;
      size_t row = (size_t)(b0 + m) * T + t;
      const float* xr = xp + row * Ntot;
      xzv[j] = xr[h];
      xrv[j] = xr[Hp + h];
      xhv[j] = xr[2 * Hp + h];
      mtv[j] = maskb ? maskb[row] : 1.0f;
      if ((h & 15) == 0 && t + 1 < T) {
        __builtin_prefetch(xr + Ntot + h, 0, 0);            // next-step z gate
        __builtin_prefetch(xr + Ntot + Hp + h, 0, 0);       // next-step r gate
        __builtin_prefetch(xr + Ntot + 2 * Hp + h, 0, 0);   // next-step h gate
      }
    }

    // ---- phase 1: rp = h @ U + br via WMMA; A-frags loaded once ----
    v16h_t af[UKS];
#pragma unroll
    for (int ks = 0; ks < UKS; ++ks) {
      v8h_t lo = *(const v8h_t*)(arow + ks * 32 + kb);
      v8h_t hi = *(const v8h_t*)(arow + ks * 32 + kb + 8);
      af[ks] = mk16(lo, hi);
    }
    for (int nt = wid; nt < NT; nt += 8) {
      float bv = brl[nt * 16 + col];
      v8f_t acc = {bv, bv, bv, bv, bv, bv, bv, bv};
#pragma unroll
      for (int ks = 0; ks < UKS; ++ks) {
        v16h_t bf = *(const v16h_t*)(usw + (size_t)(nt * UKS + ks) * 512 + lane * 16);
        acc = wmma16(af[ks], bf, acc);
      }
      int n0 = nt * 16 + col;
#pragma unroll
      for (int r = 0; r < 8; ++r) stage[(mb + r) * Ntot + n0] = acc[r];
    }
    __syncthreads();

    // ---- phase 2: fuse gates, update h-state ----
#pragma unroll
    for (int j = 0; j < EPT; ++j) {
      int e = tid + j * WGSZ;
      int m = e / Hp;
      int h = e - m * Hp;
      size_t row = (size_t)(b0 + m) * T + t;
      float rz = stage[m * Ntot + h];
      float rr = stage[m * Ntot + Hp + h];
      float rh = stage[m * Ntot + 2 * Hp + h];
      float hp = (float)hl[e];
      float z, r;
      if (actmode == 0) {
        z = hsig_f(xzv[j] + rz);
        r = hsig_f(xrv[j] + rr);
      } else {
        z = fast_sigmoid(xzv[j] + rz);
        r = fast_sigmoid(xrv[j] + rr);
      }
      float pre = xhv[j] + r * rh;
      float hh = (actmode == 0) ? fast_tanh(pre) : pre;
      float hn = z * hp + (1.f - z) * hh;
      hn = (mtv[j] > 0.5f) ? hn : hp;
      hl[e] = (_Float16)hn;
      if (hseq) hseq[row * Hp + h] = (_Float16)hn;
    }
    __syncthreads();
  }

  if (hfin) {
#pragma unroll
    for (int j = 0; j < EPT; ++j) {
      int e = tid + j * WGSZ;
      int m = e / Hp;
      int h = e - m * Hp;
      hfin[(size_t)(b0 + m) * Hp + h] = (float)hl[e];
    }
  }
}

// ---------------------------------------------------------------------------
// Sampling: z = z_mean + exp(0.5*z_log_var)*eps  (padded to 32)
// ---------------------------------------------------------------------------
__global__ void vae_sample(const float* __restrict__ zm, const float* __restrict__ zlv,
                           const float* __restrict__ eps, float* __restrict__ z, int B) {
  int i = blockIdx.x * WGSZ + threadIdx.x;
  if (i >= B * 32) return;
  int b = i >> 5, l = i & 31;
  z[i] = (l < 30)
             ? zm[i] + __builtin_amdgcn_exp2f(0.5f * LOG2E_F * zlv[i]) * eps[b * 30 + l]
             : 0.f;
}

// dec_in = concat(repeat(z), tin2) * masks  -> f16 (B*T,32)
__global__ void vae_build_decin(const float* __restrict__ z, const float* __restrict__ tin2,
                                const float* __restrict__ masks, _Float16* __restrict__ din,
                                int BT, int T) {
  int i = blockIdx.x * WGSZ + threadIdx.x;
  if (i >= BT) return;
  int b = i / T;
  const float* mk = masks + (size_t)i * 32;
  _Float16* o = din + (size_t)i * 32;
#pragma unroll
  for (int c = 0; c < 30; ++c) o[c] = (_Float16)(z[b * 32 + c] * mk[c]);
  o[30] = (_Float16)(tin2[(size_t)i * 2 + 0] * mk[30]);
  o[31] = (_Float16)(tin2[(size_t)i * 2 + 1] * mk[31]);
}

// recon = tanh(h6 @ Wd + bd) * dec_masks
__global__ void vae_dense_out(const _Float16* __restrict__ h6, const float* __restrict__ Wd,
                              const float* __restrict__ bd, const float* __restrict__ dmask,
                              float* __restrict__ out, int BT) {
  int i = blockIdx.x * WGSZ + threadIdx.x;
  if (i >= BT) return;
  const _Float16* row = h6 + (size_t)i * 192;
  float s = bd[0];
  for (int k = 0; k < 175; ++k) s += (float)row[k] * Wd[k];
  out[i] = fast_tanh(s) * dmask[i];
}

// ---------------------------------------------------------------------------
// Host orchestration
// ---------------------------------------------------------------------------
extern "C" void kernel_launch(void* const* d_in, const int* in_sizes, int n_in,
                              void* d_out, int out_size, void* d_ws, size_t ws_size,
                              hipStream_t stream) {
  (void)in_sizes; (void)n_in; (void)out_size; (void)ws_size;
  const int B = 128, T = 2048, BT = B * T;

  const float* x     = (const float*)d_in[0];
  const float* tin2  = (const float*)d_in[1];
  const float* masks = (const float*)d_in[2];
  const float* dmask = (const float*)d_in[3];
  const float* eps   = (const float*)d_in[4];
  const float* Wm[6], *Um[6], *bm[6];
  for (int l = 0; l < 6; ++l) {
    Wm[l] = (const float*)d_in[5 + 3 * l];
    Um[l] = (const float*)d_in[6 + 3 * l];
    bm[l] = (const float*)d_in[7 + 3 * l];
  }
  const float* Wd = (const float*)d_in[23];
  const float* bd = (const float*)d_in[24];
  float* out = (float*)d_out;

  // Workspace bump allocator (256B aligned)
  char* p = (char*)d_ws;
  auto alloc = [&](size_t bytes) -> char* {
    char* r = p; p += (bytes + 255) & ~(size_t)255; return r;
  };
  _Float16* Xp   = (_Float16*)alloc((size_t)BT * 32 * 2);
  float*    mb   = (float*)alloc((size_t)BT * 4);
  float*    XP   = (float*)alloc((size_t)BT * 576 * 4);
  _Float16* h1   = (_Float16*)alloc((size_t)BT * 192 * 2);
  _Float16* h2   = (_Float16*)alloc((size_t)BT * 160 * 2);
  _Float16* h5   = (_Float16*)alloc((size_t)BT * 160 * 2);
  _Float16* h6   = (_Float16*)alloc((size_t)BT * 192 * 2);
  _Float16* din  = (_Float16*)alloc((size_t)BT * 32 * 2);
  _Float16* Wsw  = (_Float16*)alloc(256 * 1024);
  _Float16* Usw  = (_Float16*)alloc(256 * 1024);
  float*    b0p  = (float*)alloc(576 * 4);
  float*    brp  = (float*)alloc(576 * 4);
  float*    zm   = (float*)alloc(B * 32 * 4);
  float*    zlv  = (float*)alloc(B * 32 * 4);
  float*    zb   = (float*)alloc(B * 32 * 4);

  // Allow >default dynamic LDS for the recurrent kernels (max 266,496 B)
  (void)hipFuncSetAttribute(reinterpret_cast<const void*>(vae_gru_recur<1>),
                            hipFuncAttributeMaxDynamicSharedMemorySize, 270336);
  (void)hipFuncSetAttribute(reinterpret_cast<const void*>(vae_gru_recur<5>),
                            hipFuncAttributeMaxDynamicSharedMemorySize, 270336);
  (void)hipFuncSetAttribute(reinterpret_cast<const void*>(vae_gru_recur<6>),
                            hipFuncAttributeMaxDynamicSharedMemorySize, 270336);

  auto run_layer = [&](const float* W, const float* U, const float* b,
                       const _Float16* inSeq, int Kp, int Krows, int H, int Hp,
                       const float* mask, _Float16* hseq, float* hfin, int actmode) {
    int Ntot = 3 * Hp, ntiles = Ntot / 16, wks = Kp / 32, uks = Hp / 32;
    int wtot = ntiles * wks * 512;
    vae_repack_frag<<<(wtot + WGSZ - 1) / WGSZ, WGSZ, 0, stream>>>(
        W, Krows, H, Hp, wks, ntiles, Wsw);
    vae_repack_bias<<<(3 * Hp + WGSZ - 1) / WGSZ, WGSZ, 0, stream>>>(b, H, Hp, b0p, brp);
    int gblocks = BT / 128;  // (BT/16 m-tiles) / 8 waves
    switch (wks) {
      case 1: vae_gemm_xp<1><<<gblocks, WGSZ, 0, stream>>>(inSeq, Wsw, b0p, XP, BT, Ntot, ntiles); break;
      case 5: vae_gemm_xp<5><<<gblocks, WGSZ, 0, stream>>>(inSeq, Wsw, b0p, XP, BT, Ntot, ntiles); break;
      default: vae_gemm_xp<6><<<gblocks, WGSZ, 0, stream>>>(inSeq, Wsw, b0p, XP, BT, Ntot, ntiles); break;
    }
    int utot = ntiles * uks * 512;
    vae_repack_frag<<<(utot + WGSZ - 1) / WGSZ, WGSZ, 0, stream>>>(
        U, H, H, Hp, uks, ntiles, Usw);
    size_t lds = (size_t)ntiles * uks * 1024 + (size_t)16 * Hp * 2 +
                 (size_t)16 * Ntot * 4 + (size_t)Ntot * 4;
    switch (uks) {
      case 1: vae_gru_recur<1><<<B / 16, WGSZ, lds, stream>>>(XP, Usw, brp, mask, hseq, hfin, T, actmode); break;
      case 5: vae_gru_recur<5><<<B / 16, WGSZ, lds, stream>>>(XP, Usw, brp, mask, hseq, hfin, T, actmode); break;
      default: vae_gru_recur<6><<<B / 16, WGSZ, lds, stream>>>(XP, Usw, brp, mask, hseq, hfin, T, actmode); break;
    }
  };

  vae_pad_x<<<BT / WGSZ, WGSZ, 0, stream>>>(x, Xp, mb, BT);

  // Encoder
  run_layer(Wm[0], Um[0], bm[0], Xp, 32, 4,   175, 192, mb, h1, nullptr, 0);
  run_layer(Wm[1], Um[1], bm[1], h1, 192, 175, 150, 160, mb, h2, nullptr, 0);
  run_layer(Wm[2], Um[2], bm[2], h2, 160, 150, 30,  32,  mb, nullptr, zm, 1);
  run_layer(Wm[3], Um[3], bm[3], h2, 160, 150, 30,  32,  mb, nullptr, zlv, 1);

  // Sampling + decoder input
  vae_sample<<<(B * 32 + WGSZ - 1) / WGSZ, WGSZ, 0, stream>>>(zm, zlv, eps, zb, B);
  vae_build_decin<<<BT / WGSZ, WGSZ, 0, stream>>>(zb, tin2, masks, din, BT, T);

  // Decoder (no sequence mask)
  run_layer(Wm[4], Um[4], bm[4], din, 32, 32,  150, 160, nullptr, h5, nullptr, 0);
  run_layer(Wm[5], Um[5], bm[5], h5, 160, 150, 175, 192, nullptr, h6, nullptr, 0);

  vae_dense_out<<<BT / WGSZ, WGSZ, 0, stream>>>(h6, Wd, bd, dmask, out, BT);
}
